// SimpleAttention_43267500540482
// MI455X (gfx1250) — matrix operand order repaired and checked by reading, not compiled, not run on hardware
//
#include <hip/hip_runtime.h>

// ---------------------------------------------------------------------------
// SimpleAttention on MI455X (gfx1250, wave32, WMMA).
//   q=xQ^T, k=xK^T, v=xV^T ; logits=qk^T+bias+mask*NEG ; w=softmax ; out=(wv)O^T
// All GEMMs via v_wmma_f32_16x16x32_bf16 (bf16 A/B, f32 accum).
// V is stored transposed (per batch) so the weights@V GEMM gets contiguous
// B-fragments; one wave owns a full 16x192 output strip so the 268 MB weights
// matrix is streamed from HBM exactly once.
// ---------------------------------------------------------------------------

typedef __bf16 bf16_t;
typedef __attribute__((ext_vector_type(16))) __bf16 v16bf;
typedef __attribute__((ext_vector_type(8)))  float  v8f;

#define D_DIM  192
#define N_SEQ  4096
#define B_SZ   4
#define ETILES (D_DIM / 16)   // 12
#define NEG_BIG (-1e9f)

// ---- WMMA fragment loaders (wave32 layouts, ISA 7.12.2) --------------------
// A 16x32 bf16: lane m=lane&15 is the row; lane half h=lane>>4 picks K-halves.
// VGPR i holds K = 16h + 2i,2i+1 (same contiguous order as B).
__device__ __forceinline__ int a_k_of(int i, int h) {
    return 16 * h + 2 * i;
}

__device__ __forceinline__ v16bf load_a_bf16(const bf16_t* __restrict__ A,
                                             int ld, int kbase, int lane) {
    const int m = lane & 15, h = lane >> 4;
    const bf16_t* row = A + (size_t)m * ld + kbase;
    v16bf a;
#pragma unroll
    for (int i = 0; i < 8; ++i) {
        const int k0 = a_k_of(i, h);
        a[2 * i]     = row[k0];
        a[2 * i + 1] = row[k0 + 1];
    }
    return a;
}

__device__ __forceinline__ v16bf load_a_f32(const float* __restrict__ A,
                                            int ld, int kbase, int lane) {
    const int m = lane & 15, h = lane >> 4;
    const float* row = A + (size_t)m * ld + kbase;
    v16bf a;
#pragma unroll
    for (int i = 0; i < 8; ++i) {
        const int k0 = a_k_of(i, h);
        a[2 * i]     = (bf16_t)row[k0];
        a[2 * i + 1] = (bf16_t)row[k0 + 1];
    }
    return a;
}

// B 32x16: lane n=lane&15 is the column; lanes 0-15 hold K=0..15, lanes 16-31
// hold K=16..31; VGPR i holds K = 16h + 2i, 2i+1.
// "bt" variants read B[k][n] = W[n*ld + k] (K contiguous in memory), so each
// fragment is a contiguous 32B (bf16) / 64B (f32) per-lane load.
__device__ __forceinline__ v16bf load_bt_f32(const float* __restrict__ W,
                                             int ld, int kbase, int lane) {
    const int n = lane & 15, h = lane >> 4;
    const float* col = W + (size_t)n * ld + kbase + 16 * h;
    v16bf b;
#pragma unroll
    for (int i = 0; i < 8; ++i) {
        b[2 * i]     = (bf16_t)col[2 * i];
        b[2 * i + 1] = (bf16_t)col[2 * i + 1];
    }
    return b;
}

__device__ __forceinline__ v16bf load_bt_bf16(const bf16_t* __restrict__ W,
                                              int ld, int kbase, int lane) {
    const int n = lane & 15, h = lane >> 4;
    const bf16_t* col = W + (size_t)n * ld + kbase + 16 * h;
    v16bf b;
#pragma unroll
    for (int i = 0; i < 8; ++i) {
        b[2 * i]     = col[2 * i];
        b[2 * i + 1] = col[2 * i + 1];
    }
    return b;
}

#define WMMA_BF16(a, b, c) \
    __builtin_amdgcn_wmma_f32_16x16x32_bf16(false, (a), false, (b), (short)0, (c), false, false)

// ---------------------------------------------------------------------------
// Kernel 1/5: Y[m,n] = sum_d X[m,d] * W[n,d]   (nn.Linear, W is [out,in])
// X f32 (M x 192), W f32 (192 x 192). One wave per 16x16 tile, K unrolled 6x.
// OUT_MODE: 0 = f32 row-major, 1 = bf16 row-major, 2 = bf16 transposed
// per batch (Yt[b][n][m_local], used for V so weights@V gets contiguous B).
// ---------------------------------------------------------------------------
template <int OUT_MODE>
__global__ void proj_gemm(const float* __restrict__ X, const float* __restrict__ W,
                          bf16_t* __restrict__ Ybf, float* __restrict__ Yf) {
    const int lane  = threadIdx.x;
    const int ntile = blockIdx.x * 16;
    const int mtile = blockIdx.y * 16;
    const float* Arow = X + (size_t)mtile * D_DIM;
    const float* Wcol = W + (size_t)ntile * D_DIM;
    v8f c = {};
#pragma unroll
    for (int kb = 0; kb < D_DIM; kb += 32) {
        v16bf a = load_a_f32(Arow, D_DIM, kb, lane);
        v16bf b = load_bt_f32(Wcol, D_DIM, kb, lane);
        c = WMMA_BF16(a, b, c);
    }
    const int n = lane & 15, h = lane >> 4;
#pragma unroll
    for (int r = 0; r < 8; ++r) {
        const int row = mtile + r + 8 * h;
        if (OUT_MODE == 2) {
            const int bb = row >> 12;            // row / N_SEQ
            const int ml = row & (N_SEQ - 1);    // row % N_SEQ
            const size_t idx = ((size_t)bb * D_DIM + ntile + n) * N_SEQ + ml;
            Ybf[idx] = (bf16_t)c[r];             // 8 consecutive bf16 per lane
        } else {
            const size_t idx = (size_t)row * D_DIM + ntile + n;
            if (OUT_MODE == 1) Ybf[idx] = (bf16_t)c[r];
            else               Yf[idx]  = c[r];
        }
    }
}

// ---------------------------------------------------------------------------
// Kernel 2/5: logits[b,q,kc] = q . k  + bias + mask*NEG  (16x64 tile per wave,
// A fragment reused across 4 column tiles). Writes into d_out weights region.
// ---------------------------------------------------------------------------
__global__ void logits_kernel(const bf16_t* __restrict__ Q, const bf16_t* __restrict__ K,
                              const float* __restrict__ bias, const float* __restrict__ mask,
                              float* __restrict__ logits) {
    const int lane  = threadIdx.x;
    const int kcol  = blockIdx.x * 64;
    const int qtile = blockIdx.y * 16;
    const int b     = blockIdx.z;
    const bf16_t* Qr = Q + ((size_t)b * N_SEQ + qtile) * D_DIM;
    const bf16_t* Kr = K + (size_t)b * N_SEQ * D_DIM;
    v8f c[4] = {{}, {}, {}, {}};
#pragma unroll
    for (int kb = 0; kb < D_DIM; kb += 32) {
        v16bf a = load_a_bf16(Qr, D_DIM, kb, lane);
#pragma unroll
        for (int j = 0; j < 4; ++j) {
            v16bf bm = load_bt_bf16(Kr + (size_t)(kcol + 16 * j) * D_DIM, D_DIM, kb, lane);
            c[j] = WMMA_BF16(a, bm, c[j]);
        }
    }
    const int n = lane & 15, h = lane >> 4;
    const size_t rowbase = (size_t)b * N_SEQ + qtile;
#pragma unroll
    for (int r = 0; r < 8; ++r) {
        const size_t grow = (rowbase + r + 8 * h) * N_SEQ;
#pragma unroll
        for (int j = 0; j < 4; ++j) {
            const size_t idx = grow + kcol + 16 * j + n;
            logits[idx] = c[j][r] + bias[idx] + NEG_BIG * mask[idx];
        }
    }
}

// ---------------------------------------------------------------------------
// Kernel 3/5: row softmax in place. One 256-thread block (8 waves) per row;
// the 4096-float row is staged in 16 KB LDS -> single HBM round trip.
// ---------------------------------------------------------------------------
__global__ void softmax_kernel(float* __restrict__ w) {
    __shared__ float buf[N_SEQ];
    __shared__ float red[8];
    const int tid = threadIdx.x;
    float* p = w + (size_t)blockIdx.x * N_SEQ;

    float m = -3.4e38f;
    for (int i = tid; i < N_SEQ; i += 256) {
        const float v = p[i];
        buf[i] = v;
        m = fmaxf(m, v);
    }
#pragma unroll
    for (int off = 16; off; off >>= 1) m = fmaxf(m, __shfl_xor(m, off, 32));
    if ((tid & 31) == 0) red[tid >> 5] = m;
    __syncthreads();
    m = red[0];
#pragma unroll
    for (int i = 1; i < 8; ++i) m = fmaxf(m, red[i]);
    __syncthreads();  // protect red[] reuse

    float s = 0.f;
    for (int i = tid; i < N_SEQ; i += 256) {
        const float e = __expf(buf[i] - m);
        buf[i] = e;
        s += e;
    }
#pragma unroll
    for (int off = 16; off; off >>= 1) s += __shfl_xor(s, off, 32);
    if ((tid & 31) == 0) red[tid >> 5] = s;
    __syncthreads();
    s = 0.f;
#pragma unroll
    for (int i = 0; i < 8; ++i) s += red[i];
    const float inv = 1.0f / s;
    for (int i = tid; i < N_SEQ; i += 256) p[i] = buf[i] * inv;
}

// ---------------------------------------------------------------------------
// Kernel 4/5: attn[b,q,:] = weights[b,q,:] @ v[b,:,:]. One wave owns a full
// 16x192 strip: 12 accumulators, one A (weights, f32->bf16 on the fly) load
// feeds 12 WMMAs per K-step; weights stream from HBM exactly once, Vt (1.5 MB
// per batch, bf16 transposed) is L2-resident for its re-reads.
// ---------------------------------------------------------------------------
__global__ void av_kernel(const float* __restrict__ Wt, const bf16_t* __restrict__ Vt,
                          float* __restrict__ Out) {
    const int lane  = threadIdx.x;
    const int qtile = blockIdx.x * 16;
    const int b     = blockIdx.y;
    const float*  A   = Wt + ((size_t)b * N_SEQ + qtile) * N_SEQ;
    const bf16_t* VtB = Vt + (size_t)b * D_DIM * N_SEQ;
    v8f c[ETILES];
#pragma unroll
    for (int e = 0; e < ETILES; ++e) c[e] = (v8f){};

    for (int kb = 0; kb < N_SEQ; kb += 32) {
        if (kb + 32 < N_SEQ)
            __builtin_prefetch(A + (size_t)(lane & 15) * N_SEQ + kb + 32, 0, 0);
        v16bf a = load_a_f32(A, N_SEQ, kb, lane);
#pragma unroll
        for (int e = 0; e < ETILES; ++e) {
            v16bf bm = load_bt_bf16(VtB + (size_t)(16 * e) * N_SEQ, N_SEQ, kb, lane);
            c[e] = WMMA_BF16(a, bm, c[e]);
        }
    }
    const int n = lane & 15, h = lane >> 4;
#pragma unroll
    for (int e = 0; e < ETILES; ++e) {
#pragma unroll
        for (int r = 0; r < 8; ++r) {
            const size_t idx =
                ((size_t)b * N_SEQ + qtile + r + 8 * h) * D_DIM + 16 * e + n;
            Out[idx] = c[e][r];
        }
    }
}

// ---------------------------------------------------------------------------
extern "C" void kernel_launch(void* const* d_in, const int* in_sizes, int n_in,
                              void* d_out, int out_size, void* d_ws, size_t ws_size,
                              hipStream_t stream) {
    (void)in_sizes; (void)n_in; (void)out_size; (void)ws_size;

    const float* query = (const float*)d_in[0];
    const float* key   = (const float*)d_in[1];
    const float* value = (const float*)d_in[2];
    const float* mask  = (const float*)d_in[3];
    const float* bias  = (const float*)d_in[4];
    const float* q_w   = (const float*)d_in[5];
    const float* k_w   = (const float*)d_in[6];
    const float* v_w   = (const float*)d_in[7];
    const float* o_w   = (const float*)d_in[8];

    const size_t nd = (size_t)B_SZ * N_SEQ * D_DIM;  // elements in q/k/v/out

    // workspace: q,k bf16 + Vt bf16 (3 * 6.3 MB) then attn f32 (12.6 MB)
    char*   ws   = (char*)d_ws;
    bf16_t* qb   = (bf16_t*)ws;
    bf16_t* kb   = qb + nd;
    bf16_t* vt   = kb + nd;            // transposed per batch: [b][e][k]
    float*  attn = (float*)(ws + 3 * nd * sizeof(bf16_t));

    float* out_final = (float*)d_out;           // (B,N,D)
    float* weights   = out_final + nd;          // (B,N,N)

    const dim3 wave(32);
    const dim3 gproj(D_DIM / 16, (B_SZ * N_SEQ) / 16);

    proj_gemm<1><<<gproj, wave, 0, stream>>>(query, q_w, qb, nullptr);
    proj_gemm<1><<<gproj, wave, 0, stream>>>(key,   k_w, kb, nullptr);
    proj_gemm<2><<<gproj, wave, 0, stream>>>(value, v_w, vt, nullptr);

    const dim3 glog(N_SEQ / 64, N_SEQ / 16, B_SZ);
    logits_kernel<<<glog, wave, 0, stream>>>(qb, kb, bias, mask, weights);

    softmax_kernel<<<dim3(B_SZ * N_SEQ), dim3(256), 0, stream>>>(weights);

    const dim3 gav(N_SEQ / 16, B_SZ);
    av_kernel<<<gav, wave, 0, stream>>>(weights, vt, attn);

    proj_gemm<0><<<gproj, wave, 0, stream>>>(attn, o_w, nullptr, out_final);
}
